// NewHeadedAttentionBlock_86844238725729
// MI455X (gfx1250) — matrix operand order repaired and checked by reading, not compiled, not run on hardware
//
#include <hip/hip_runtime.h>
#include <hip/hip_bf16.h>

typedef __bf16 bf16_t;
typedef __attribute__((ext_vector_type(16))) __bf16 v16bf;
typedef __attribute__((ext_vector_type(8)))  float  v8f;

#define MODEL_DIM 256
#define NUM_HEADS 8
#define HEAD_DIM  32
#define BATCH     4
#define SEQ       2048
#define ROWS      (BATCH * SEQ)   // 8192

// ---------------------------------------------------------------------------
// WMMA fragment helpers (wave32; this code uses contiguous K order k = 16*(lane>>4) + s)
// ---------------------------------------------------------------------------

// A-matrix 16x32 bf16: lane holds row m = lane&15; K elements
//   vec[0..7]  = K  2*kbase..2*kbase+7      (kbase = (lane>>4)*8)
//   vec[8..15] = K  2*kbase+8..2*kbase+15
__device__ __forceinline__ v16bf load_afrag_bf16(const bf16_t* __restrict__ row, int kbase) {
    union { uint4 u[2]; v16bf v; } t;
    t.u[0] = *reinterpret_cast<const uint4*>(row + 2 * kbase);
    t.u[1] = *reinterpret_cast<const uint4*>(row + 2 * kbase + 8);
    return t.v;
}

// B-matrix 32x16 bf16: lane holds col n = lane&15; K elements
//   vec[0..15] = K kbB..kbB+15  (kbB = lane<16 ? 0 : 16), contiguous in memory
// Caller passes pointer already offset by kbB along K.
__device__ __forceinline__ v16bf load_bfrag_bf16(const bf16_t* __restrict__ p) {
    union { uint4 u[2]; v16bf v; } t;
    t.u[0] = *reinterpret_cast<const uint4*>(p);
    t.u[1] = *reinterpret_cast<const uint4*>(p + 8);
    return t.v;
}

// Same fragments built from f32 memory with on-the-fly bf16 conversion.
__device__ __forceinline__ v16bf cvt_afrag_f32(const float* __restrict__ row, int kbase) {
    v16bf v;
#pragma unroll
    for (int i = 0; i < 8; ++i) v[i] = (bf16_t)row[2 * kbase + i];
#pragma unroll
    for (int i = 0; i < 8; ++i) v[8 + i] = (bf16_t)row[2 * kbase + 8 + i];
    return v;
}
__device__ __forceinline__ v16bf cvt_bfrag_f32(const float* __restrict__ p) {
    v16bf v;
#pragma unroll
    for (int i = 0; i < 16; ++i) v[i] = (bf16_t)p[i];
    return v;
}

__device__ __forceinline__ v8f wmma_bf16(v16bf a, v16bf b, v8f c) {
    // (neg_a, A, neg_b, B, c_mod, C, reuse_a, reuse_b)
    return __builtin_amdgcn_wmma_f32_16x16x32_bf16(false, a, false, b, (short)0, c,
                                                   false, false);
}

// ---------------------------------------------------------------------------
// Kernel 1: fused Q/K/V projection  out = X @ W^T + b, bf16 outputs.
//   Q stored [B*H][S][32] pre-scaled by 1/sqrt(256)=1/16
//   K stored [B*H][S][32]
//   V stored transposed [B*H][32][S]  (so attention P*V B-frags are contiguous)
// One wave per 16x16 tile. waves = 512 rowtiles * 3 mats * 16 coltiles = 24576.
// ---------------------------------------------------------------------------
__global__ __launch_bounds__(256) void qkv_proj_kernel(
    const float* __restrict__ X,
    const float* __restrict__ Wq, const float* __restrict__ bq,
    const float* __restrict__ Wk, const float* __restrict__ bk,
    const float* __restrict__ Wv, const float* __restrict__ bv,
    bf16_t* __restrict__ Qh, bf16_t* __restrict__ Kh, bf16_t* __restrict__ Vt)
{
    const int wave = (int)((blockIdx.x * blockDim.x + threadIdx.x) >> 5);
    const int lane = (int)(threadIdx.x & 31);
    const int ct   = wave & 15;
    const int tmp  = wave >> 4;
    const int mat  = tmp % 3;           // 0=Q 1=K 2=V
    const int rt   = tmp / 3;           // row tile 0..511
    if (rt >= ROWS / 16) return;

    const float *W, *bias;
    if (mat == 0)      { W = Wq; bias = bq; }
    else if (mat == 1) { W = Wk; bias = bk; }
    else               { W = Wv; bias = bv; }

    const int n     = lane & 15;
    const int kbase = (lane >> 4) * 8;          // A-frag K base
    const int kbB   = (lane < 16) ? 0 : 16;     // B-frag K base
    const int col   = ct * 16 + n;

    const float* xrow = X + (size_t)(rt * 16 + (lane & 15)) * MODEL_DIM;
    const float* wrow = W + (size_t)col * MODEL_DIM + kbB;

    v8f acc = {};
#pragma unroll
    for (int kb = 0; kb < MODEL_DIM / 32; ++kb) {
        v16bf a = cvt_afrag_f32(xrow + kb * 32, kbase);
        v16bf b = cvt_bfrag_f32(wrow + kb * 32);
        acc = wmma_bf16(a, b, acc);
    }

    const float bcol  = bias[col];
    const int   mbase = (lane >> 4) * 8;        // C-layout M base
    const int   h     = col >> 5;
    const int   hd    = col & 31;
    const int   row0  = rt * 16;
    const int   bidx  = row0 >> 11;             // batch (tiles never cross batch)
    const int   pair  = bidx * NUM_HEADS + h;
    const int   s0    = row0 & (SEQ - 1);

    if (mat == 2) {
        // transposed V: lane writes 8 consecutive s positions -> one b128 store
        bf16_t* dst = Vt + ((size_t)pair * HEAD_DIM + hd) * SEQ + s0 + mbase;
        union { uint4 u; bf16_t e[8]; } pk;
#pragma unroll
        for (int j = 0; j < 8; ++j) pk.e[j] = (bf16_t)(acc[j] + bcol);
        *reinterpret_cast<uint4*>(dst) = pk.u;
    } else {
        const float scale = (mat == 0) ? 0.0625f : 1.0f;   // 1/sqrt(MODEL_DIM)
        bf16_t* base = (mat == 0 ? Qh : Kh) +
                       ((size_t)pair * SEQ + s0 + mbase) * HEAD_DIM + hd;
#pragma unroll
        for (int j = 0; j < 8; ++j)
            base[(size_t)j * HEAD_DIM] = (bf16_t)((acc[j] + bcol) * scale);
    }
}

// ---------------------------------------------------------------------------
// Kernel 2: flash-attention. One wave = 16 queries x full head (hd=32).
// HEAD_DIM==32 == WMMA K-depth, so Q.K^T is ONE wmma per 16x16 score tile.
// waves = 32 (b,h) pairs * 128 qtiles = 4096; 4 waves/block.
// ---------------------------------------------------------------------------
__global__ __launch_bounds__(128) void attn_kernel(
    const bf16_t* __restrict__ Qh, const bf16_t* __restrict__ Kh,
    const bf16_t* __restrict__ Vt, const unsigned char* __restrict__ qmask,
    bf16_t* __restrict__ AO)
{
    __shared__ __attribute__((aligned(16))) bf16_t Pld[4][16][32]; // per-wave P tile

    const int wid  = (int)(threadIdx.x >> 5);
    const int lane = (int)(threadIdx.x & 31);
    const int wave = (int)(blockIdx.x * 4 + wid);
    const int qt   = wave & 127;        // query tile
    const int pair = wave >> 7;         // b*8 + h
    const int bidx = pair >> 3;
    const int h    = pair & 7;

    const int n     = lane & 15;
    const int kbase = (lane >> 4) * 8;      // A-frag K base == C-layout M base
    const int kbB   = (lane < 16) ? 0 : 16; // B-frag K base
    const int mbase = kbase;

    // Q fragment (already scaled by 1/16 at projection time)
    const bf16_t* qrow = Qh + ((size_t)pair * SEQ + qt * 16 + (lane & 15)) * HEAD_DIM;
    const v16bf qf = load_afrag_bf16(qrow, kbase);

    const bf16_t* kmat  = Kh + (size_t)pair * SEQ * HEAD_DIM;
    const bf16_t* vtmat = Vt + (size_t)pair * HEAD_DIM * SEQ;
    const unsigned char* km = qmask + (size_t)bidx * SEQ;

    v8f o0 = {}, o1 = {};
    float mrun[8], ssum[8];
#pragma unroll
    for (int j = 0; j < 8; ++j) { mrun[j] = -__builtin_inff(); ssum[j] = 0.f; }

    bf16_t* pl = &Pld[wid][0][0];
    const v8f zc = {};

    for (int kt = 0; kt < SEQ; kt += 32) {
        // K^T fragments: B[e][key] = K[key][e]; lane n = key, contiguous in e
        v16bf kf0 = load_bfrag_bf16(kmat + (size_t)(kt + n)      * HEAD_DIM + kbB);
        v16bf kf1 = load_bfrag_bf16(kmat + (size_t)(kt + 16 + n) * HEAD_DIM + kbB);
        v8f s0 = wmma_bf16(qf, kf0, zc);
        v8f s1 = wmma_bf16(qf, kf1, zc);

        const bool msk0 = km[kt + n] != 0;
        const bool msk1 = km[kt + 16 + n] != 0;

        if (kt + 32 < SEQ) {  // overlap next tile fetch with softmax math
            __builtin_prefetch(kmat  + (size_t)(kt + 32 + n) * HEAD_DIM, 0, 3);
            __builtin_prefetch(vtmat + (size_t)n * SEQ + kt + 32,        0, 3);
        }

        // online softmax, per C-layout row M = mbase+j (lives in one 16-lane half)
#pragma unroll
        for (int j = 0; j < 8; ++j) {
            float a0 = msk0 ? s0[j] : -__builtin_inff();
            float a1 = msk1 ? s1[j] : -__builtin_inff();
            float t = fmaxf(a0, a1);
            t = fmaxf(t, __shfl_xor(t, 1, 32));
            t = fmaxf(t, __shfl_xor(t, 2, 32));
            t = fmaxf(t, __shfl_xor(t, 4, 32));
            t = fmaxf(t, __shfl_xor(t, 8, 32));
            float mnew = fmaxf(mrun[j], t);
            float p0, p1, scl;
            if (mnew == -__builtin_inff()) { p0 = 0.f; p1 = 0.f; scl = 1.f; }
            else {
                scl = __expf(mrun[j] - mnew);
                p0  = __expf(a0 - mnew);
                p1  = __expf(a1 - mnew);
            }
            mrun[j] = mnew;
            float rs = p0 + p1;
            rs += __shfl_xor(rs, 1, 32);
            rs += __shfl_xor(rs, 2, 32);
            rs += __shfl_xor(rs, 4, 32);
            rs += __shfl_xor(rs, 8, 32);
            ssum[j] = ssum[j] * scl + rs;
            o0[j] *= scl;
            o1[j] *= scl;
            // stage P (C-layout) into LDS, row = query, col = local key 0..31
            pl[(mbase + j) * 32 + n]      = (bf16_t)p0;
            pl[(mbase + j) * 32 + 16 + n] = (bf16_t)p1;
        }

        // same-wave LDS is in-order; fence the compiler + DS counter anyway
        __asm__ volatile("s_wait_dscnt 0x0" ::: "memory");

        // re-read P in A-fragment layout (2x ds_load_b128 per lane)
        v16bf pf = load_afrag_bf16(pl + (lane & 15) * 32, kbase);

        // V fragments from transposed V: B[key][hd] = Vt[hd][key], contiguous in key
        v16bf vf0 = load_bfrag_bf16(vtmat + (size_t)n        * SEQ + kt + kbB);
        v16bf vf1 = load_bfrag_bf16(vtmat + (size_t)(16 + n) * SEQ + kt + kbB);
        o0 = wmma_bf16(pf, vf0, o0);
        o1 = wmma_bf16(pf, vf1, o1);
    }

    // normalize and store attn output as bf16 [B][S][256] (head-concat layout)
    bf16_t* aob = AO + ((size_t)bidx * SEQ + qt * 16) * MODEL_DIM + h * HEAD_DIM;
#pragma unroll
    for (int j = 0; j < 8; ++j) {
        float inv = (ssum[j] > 0.f) ? (1.0f / ssum[j]) : 0.f;
        aob[(size_t)(mbase + j) * MODEL_DIM + n]      = (bf16_t)(o0[j] * inv);
        aob[(size_t)(mbase + j) * MODEL_DIM + 16 + n] = (bf16_t)(o1[j] * inv);
    }
}

// ---------------------------------------------------------------------------
// Kernel 3: output projection  out = AO @ Wo^T + bo (f32 out), q-mask zeroing.
// waves = 512 rowtiles * 16 coltiles = 8192; 8 waves/block.
// ---------------------------------------------------------------------------
__global__ __launch_bounds__(256) void out_proj_kernel(
    const bf16_t* __restrict__ AO, const float* __restrict__ Wo,
    const float* __restrict__ bo, const unsigned char* __restrict__ qmask,
    float* __restrict__ out)
{
    const int wave = (int)((blockIdx.x * blockDim.x + threadIdx.x) >> 5);
    const int lane = (int)(threadIdx.x & 31);
    const int ct = wave & 15;
    const int rt = wave >> 4;
    if (rt >= ROWS / 16) return;

    const int n     = lane & 15;
    const int kbase = (lane >> 4) * 8;
    const int kbB   = (lane < 16) ? 0 : 16;
    const int col   = ct * 16 + n;

    const bf16_t* arow = AO + (size_t)(rt * 16 + (lane & 15)) * MODEL_DIM;
    const float*  wrow = Wo + (size_t)col * MODEL_DIM + kbB;

    v8f acc = {};
#pragma unroll
    for (int kb = 0; kb < MODEL_DIM / 32; ++kb) {
        v16bf a = load_afrag_bf16(arow + kb * 32, kbase);
        v16bf b = cvt_bfrag_f32(wrow + kb * 32);
        acc = wmma_bf16(a, b, acc);
    }

    const float bcol  = bo[col];
    const int   mbase = kbase;
    const int   row0  = rt * 16;
#pragma unroll
    for (int j = 0; j < 8; ++j) {
        const int row = row0 + mbase + j;
        float v = acc[j] + bcol;
        if (!qmask[row]) v = 0.f;   // masked q rows -> nan_to_num(NaN) == 0
        out[(size_t)row * MODEL_DIM + col] = v;
    }
}

// ---------------------------------------------------------------------------
extern "C" void kernel_launch(void* const* d_in, const int* in_sizes, int n_in,
                              void* d_out, int out_size, void* d_ws, size_t ws_size,
                              hipStream_t stream)
{
    const float*         q     = (const float*)d_in[0];
    const unsigned char* qmask = (const unsigned char*)d_in[1];
    const float* Wq = (const float*)d_in[2];
    const float* bq = (const float*)d_in[3];
    const float* Wk = (const float*)d_in[4];
    const float* bk = (const float*)d_in[5];
    const float* Wv = (const float*)d_in[6];
    const float* bv = (const float*)d_in[7];
    const float* Wo = (const float*)d_in[8];
    const float* bo = (const float*)d_in[9];
    float* out = (float*)d_out;

    const size_t nBH = (size_t)BATCH * NUM_HEADS * SEQ * HEAD_DIM; // 2,097,152 elems
    bf16_t* Qh = (bf16_t*)d_ws;
    bf16_t* Kh = Qh + nBH;
    bf16_t* Vt = Kh + nBH;
    bf16_t* AO = Vt + nBH;   // ROWS*MODEL_DIM == nBH elems; total ws = 16 MB

    // 512 rowtiles * 3 mats * 16 coltiles = 24576 waves / 8 per block
    qkv_proj_kernel<<<3072, 256, 0, stream>>>(q, Wq, bq, Wk, bk, Wv, bv, Qh, Kh, Vt);
    // 32 pairs * 128 qtiles = 4096 waves / 4 per block
    attn_kernel<<<1024, 128, 0, stream>>>(Qh, Kh, Vt, qmask, AO);
    // 512 rowtiles * 16 coltiles = 8192 waves / 8 per block
    out_proj_kernel<<<1024, 256, 0, stream>>>(AO, Wo, bo, qmask, out);
}
